// ConditionalModel2_87058987090621
// MI455X (gfx1250) — matrix operand order repaired and checked by reading, not compiled, not run on hardware
//
#include <hip/hip_runtime.h>

// ---------------- problem constants (from reference file) ----------------
#define B_       2
#define H_       32
#define S_NEW_   2048
#define D_       128
#define MAX_SEQ_ 8192
#define START_I  4096
#define END_I    6144     // K length for attention

// ---------------- tiling ----------------
#define QTILE   128           // q rows per block (8 waves x 16)
#define CHUNK   64            // K rows staged per iteration (4 score tiles)
#define KROW    (D_ + 8)      // padded f16 row stride, K tile   (136 -> 272B, 16B-mult)
#define VROW    (CHUNK + 8)   // padded f16 row stride, V^T tile (72  -> 144B, 16B-mult)

// scores computed directly in log2 units: scale * log2(e) folded into Q
#define SC2     (0.08838834764831845f * 1.4426950408889634f)

typedef __attribute__((ext_vector_type(16))) _Float16 v16h;
typedef __attribute__((ext_vector_type(2)))  _Float16 h2;
typedef __attribute__((ext_vector_type(8)))  float    v8f;

union Frag16 {                // one WMMA f16 operand (16 f16 / lane = 8 VGPRs)
  v16h v;
  _Float16 h[16];
  unsigned int u[8];
  uint4 q[2];
};

#if __has_builtin(__builtin_amdgcn_cvt_pkrtz)
static __device__ __forceinline__ unsigned int pack_f16x2(float a, float b) {
  auto p = __builtin_amdgcn_cvt_pkrtz(a, b);   // v_cvt_pkrtz_f16_f32, 1 VALU op
  return __builtin_bit_cast(unsigned int, p);
}
#else
static __device__ __forceinline__ unsigned int pack_f16x2(float a, float b) {
  h2 p;
  p.x = (_Float16)a;
  p.y = (_Float16)b;
  return __builtin_bit_cast(unsigned int, p);
}
#endif

#if __has_builtin(__builtin_amdgcn_exp2f)
#define EXP2F(x) __builtin_amdgcn_exp2f(x)
#else
#define EXP2F(x) exp2f(x)
#endif

static __device__ __forceinline__ v8f vzero8() {
  v8f z;
  for (int i = 0; i < 8; ++i) z[i] = 0.0f;
  return z;
}

// ---------------------------------------------------------------------------
// Kernel 1: build new_k1 / new_v1 (splice k,v into k1,v1 at [START_I, END_I))
// ---------------------------------------------------------------------------
__global__ void splice_kv(const float* __restrict__ k,
                          const float* __restrict__ v,
                          const float* __restrict__ k1,
                          const float* __restrict__ v1,
                          float* __restrict__ nk1,
                          float* __restrict__ nv1) {
  long i = (long)blockIdx.x * blockDim.x + threadIdx.x;  // float4 index
  int  d4 = (int)(i & 31);              // D/4 = 32
  long t  = i >> 5;
  int  s  = (int)(t & (MAX_SEQ_ - 1));  // 8192
  long bh = t >> 13;

  float4 kv, vv;
  if (s >= START_I && s < END_I) {
    long idx = (bh * S_NEW_ + (s - START_I)) * 32 + d4;
    kv = ((const float4*)k)[idx];
    vv = ((const float4*)v)[idx];
  } else {
    kv = ((const float4*)k1)[i];
    vv = ((const float4*)v1)[i];
  }
  ((float4*)nk1)[i] = kv;
  ((float4*)nv1)[i] = vv;
}

// ---------------------------------------------------------------------------
// Kernel 2: transposed-domain flash attention, v_wmma_f32_16x16x32_f16
//   S^T = K Q^T   (lane column = q row; Q pre-scaled by scale*log2e)
//   O^T = V^T P^T (P^T in-lane from S^T output; l via ones-matrix WMMA)
//   grid = (S_NEW/QTILE, B*H), block = 256 (8 wave32)
// ---------------------------------------------------------------------------
__launch_bounds__(256)
__global__ void flash_attn(const float* __restrict__ q,
                           const float* __restrict__ kc,   // new_k1
                           const float* __restrict__ vc,   // new_v1
                           float* __restrict__ out) {
  __shared__ __align__(16) unsigned short k_lds[CHUNK * KROW];  // 64 x 136 f16
  __shared__ __align__(16) unsigned short v_lds[D_ * VROW];     // 128 x 72 f16 (transposed)

  const int qtile = blockIdx.x;
  const int bh    = blockIdx.y;
  const int tid   = threadIdx.x;
  const int wave  = tid >> 5;
  const int lane  = tid & 31;
  const int half  = lane >> 4;   // wave32 half
  const int ln    = lane & 15;

  const float* qb = q  + (long)bh * S_NEW_   * D_;
  const float* kb = kc + (long)bh * MAX_SEQ_ * D_;
  const float* vb = vc + (long)bh * MAX_SEQ_ * D_;
  float*       ob = out + (long)bh * S_NEW_  * D_;

  // ---- Q^T B-fragments (loop invariant): lane -> q row; pre-scaled by SC2 ----
  const int qrow0 = qtile * QTILE + wave * 16;
  Frag16 bq[4];
  {
    const float* qr = qb + (long)(qrow0 + ln) * D_;
    for (int ds = 0; ds < 4; ++ds)
      for (int i = 0; i < 8; ++i) {
        const int d = ds * 32 + half * 16 + 2 * i;
        bq[ds].u[i] = pack_f16x2(qr[d] * SC2, qr[d + 1] * SC2);
      }
  }

  // ones A-matrix for the row-sum WMMA (1.0f16 = 0x3C00)
  Frag16 fones;
  for (int i = 0; i < 8; ++i) fones.u[i] = 0x3C003C00u;

  v8f acc[8];                         // O^T tiles: M = d (tile j), N = q
  for (int j = 0; j < 8; ++j) acc[j] = vzero8();
  v8f accl = vzero8();                // l accumulator (only element 0 is read)
  float m_i = -1e30f;                 // running max (log2 units), per-lane scalar

  // staging: each thread owns a PAIR of k rows x 16 cols (k-adjacent f16 packing)
  const int prow = tid >> 3;          // 0..31 -> rows 2p, 2p+1
  const int pcol = (tid & 7) << 4;    // 0,16,...,112

  for (int c0 = 0; c0 < END_I; c0 += CHUNK) {
    __syncthreads();  // previous chunk's LDS reads complete
    {
      const int r0 = 2 * prow, r1 = 2 * prow + 1;
      const float4* ka  = (const float4*)(kb + (long)(c0 + r0) * D_ + pcol);
      const float4* kb4 = (const float4*)(kb + (long)(c0 + r1) * D_ + pcol);
      const float4* va  = (const float4*)(vb + (long)(c0 + r0) * D_ + pcol);
      const float4* vb4 = (const float4*)(vb + (long)(c0 + r1) * D_ + pcol);

      // K rows: row-major f16, 2x (16 floats -> 2 uint4)
      {
        uint4* d0 = (uint4*)&k_lds[r0 * KROW + pcol];
        uint4* d1 = (uint4*)&k_lds[r1 * KROW + pcol];
        float4 x0 = ka[0], x1 = ka[1], x2 = ka[2], x3 = ka[3];
        uint4 t0, t1;
        t0.x = pack_f16x2(x0.x, x0.y); t0.y = pack_f16x2(x0.z, x0.w);
        t0.z = pack_f16x2(x1.x, x1.y); t0.w = pack_f16x2(x1.z, x1.w);
        t1.x = pack_f16x2(x2.x, x2.y); t1.y = pack_f16x2(x2.z, x2.w);
        t1.z = pack_f16x2(x3.x, x3.y); t1.w = pack_f16x2(x3.z, x3.w);
        d0[0] = t0; d0[1] = t1;
        x0 = kb4[0]; x1 = kb4[1]; x2 = kb4[2]; x3 = kb4[3];
        t0.x = pack_f16x2(x0.x, x0.y); t0.y = pack_f16x2(x0.z, x0.w);
        t0.z = pack_f16x2(x1.x, x1.y); t0.w = pack_f16x2(x1.z, x1.w);
        t1.x = pack_f16x2(x2.x, x2.y); t1.y = pack_f16x2(x2.z, x2.w);
        t1.z = pack_f16x2(x3.x, x3.y); t1.w = pack_f16x2(x3.z, x3.w);
        d1[0] = t0; d1[1] = t1;
      }
      // V transposed: v_lds[d][k]; pack (row r0, row r1) -> one b32 per d
      for (int g = 0; g < 4; ++g) {
        float4 a = va[g], b = vb4[g];
        const int dbase = pcol + g * 4;
        *(unsigned int*)&v_lds[(dbase + 0) * VROW + r0] = pack_f16x2(a.x, b.x);
        *(unsigned int*)&v_lds[(dbase + 1) * VROW + r0] = pack_f16x2(a.y, b.y);
        *(unsigned int*)&v_lds[(dbase + 2) * VROW + r0] = pack_f16x2(a.z, b.z);
        *(unsigned int*)&v_lds[(dbase + 3) * VROW + r0] = pack_f16x2(a.w, b.w);
      }
    }
    __syncthreads();

    // ---- S^T tiles: st[t] = K[16t..16t+15] . Q^T  (contraction over D) ----
    v8f st[4];
    for (int t = 0; t < 4; ++t) st[t] = vzero8();
    for (int ds = 0; ds < 4; ++ds) {
      for (int t = 0; t < 4; ++t) {
        Frag16 ak;  // A: lane -> k row; slots: e -> d=16h+e
        const unsigned short* kr = &k_lds[(t * 16 + ln) * KROW + ds * 32];
        ak.q[0] = *(const uint4*)(kr + half * 16);
        ak.q[1] = *(const uint4*)(kr + half * 16 + 8);
        st[t] = __builtin_amdgcn_wmma_f32_16x16x32_f16(false, ak.v, false, bq[ds].v,
                                                       (short)0, st[t], false, false);
      }
    }

    // ---- online softmax (log2 units; lane scalar stats for q = qrow0+ln) ----
    float tmax = -1e30f;
    for (int t = 0; t < 4; ++t)
      for (int r = 0; r < 8; ++r)
        tmax = fmaxf(tmax, st[t][r]);
    tmax = fmaxf(tmax, __shfl_xor(tmax, 16, 32));   // combine wave halves
    const float nm   = fmaxf(m_i, tmax);
    const float corr = EXP2F(m_i - nm);
    m_i = nm;

    Frag16 bp0, bp1;  // P^T B-fragments: slot e<8 <- tile (2c), e>=8 <- tile (2c+1)
    for (int t = 0; t < 4; ++t) {
      Frag16& bp = (t < 2) ? bp0 : bp1;
      const int ub = (t & 1) * 4;
      for (int r = 0; r < 8; r += 2) {
        float p0 = EXP2F(st[t][r]     - nm);
        float p1 = EXP2F(st[t][r + 1] - nm);
        bp.u[ub + (r >> 1)] = pack_f16x2(p0, p1);
      }
    }

    for (int j = 0; j < 8; ++j)
      for (int r = 0; r < 8; ++r)
        acc[j][r] *= corr;
    accl[0] *= corr;   // only row 0 of the ones-product is ever read

    // l += 1 . P^T  (row-sum via WMMA, co-executes on the matrix pipe)
    accl = __builtin_amdgcn_wmma_f32_16x16x32_f16(false, fones.v, false, bp0.v,
                                                  (short)0, accl, false, false);
    accl = __builtin_amdgcn_wmma_f32_16x16x32_f16(false, fones.v, false, bp1.v,
                                                  (short)0, accl, false, false);

    // ---- O^T += V^T P^T : A reads in table order (e<8 -> k=8h+e, e>=8 -> k=16+8h+(e-8)) ----
    for (int j = 0; j < 8; ++j) {
      const unsigned short* vr = &v_lds[(j * 16 + ln) * VROW];
      Frag16 av0, av1;
      av0.q[0] = *(const uint4*)(vr + half * 8);
      av0.q[1] = *(const uint4*)(vr + 16 + half * 8);
      av1.q[0] = *(const uint4*)(vr + 32 + half * 8);
      av1.q[1] = *(const uint4*)(vr + 48 + half * 8);
      acc[j] = __builtin_amdgcn_wmma_f32_16x16x32_f16(false, av0.v, false, bp0.v,
                                                      (short)0, acc[j], false, false);
      acc[j] = __builtin_amdgcn_wmma_f32_16x16x32_f16(false, av1.v, false, bp1.v,
                                                      (short)0, acc[j], false, false);
    }
  }

  // ---- normalize and write: lane -> q row; acc[j][r] -> d = 16j + 8h + r ----
  const float inv = 1.0f / accl[0];
  float* orow = ob + (long)(qrow0 + ln) * D_;
  for (int j = 0; j < 8; ++j) {
    float4 o0, o1;
    o0.x = acc[j][0] * inv; o0.y = acc[j][1] * inv;
    o0.z = acc[j][2] * inv; o0.w = acc[j][3] * inv;
    o1.x = acc[j][4] * inv; o1.y = acc[j][5] * inv;
    o1.z = acc[j][6] * inv; o1.w = acc[j][7] * inv;
    float* dst = orow + j * 16 + half * 8;
    *(float4*)(dst)     = o0;
    *(float4*)(dst + 4) = o1;
  }
}

// ---------------------------------------------------------------------------
extern "C" void kernel_launch(void* const* d_in, const int* in_sizes, int n_in,
                              void* d_out, int out_size, void* d_ws, size_t ws_size,
                              hipStream_t stream) {
  const float* q  = (const float*)d_in[0];
  const float* k  = (const float*)d_in[1];
  const float* v  = (const float*)d_in[2];
  const float* k1 = (const float*)d_in[3];
  const float* v1 = (const float*)d_in[4];
  (void)in_sizes; (void)n_in; (void)out_size; (void)d_ws; (void)ws_size;

  float* out = (float*)d_out;
  float* nk1 = out + (long)B_ * H_ * S_NEW_ * D_;     // 16,777,216
  float* nv1 = nk1 + (long)B_ * H_ * MAX_SEQ_ * D_;   // +67,108,864

  // 1) splice caches straight into d_out (attention reads them stream-ordered)
  long n4 = (long)B_ * H_ * MAX_SEQ_ * (D_ / 4);      // 16,777,216 float4 slots
  splice_kv<<<dim3((unsigned)(n4 / 256)), dim3(256), 0, stream>>>(k, v, k1, v1, nk1, nv1);

  // 2) flash attention over K = first END_I rows of the spliced caches
  dim3 grid(S_NEW_ / QTILE, B_ * H_);
  flash_attn<<<grid, dim3(256), 0, stream>>>(q, nk1, nv1, out);
}
